// ChannelDeAttention_40381282517522
// MI455X (gfx1250) — matrix operand order repaired and checked by reading, not compiled, not run on hardware
//
#include <hip/hip_runtime.h>
#include <hip/hip_bf16.h>

typedef __bf16 bf16_t;
typedef __attribute__((ext_vector_type(16))) __bf16 v16bf;
typedef __attribute__((ext_vector_type(8)))  __bf16 v8bf;
typedef __attribute__((ext_vector_type(8)))  float  v8f;

#define FLAG_OUT_BF16 1
#define FLAG_TRANS    2
#define FLAG_BIAS     4

union Frag32 { v16bf v; v8bf h[2]; };

// ---------------- transpose x [B,512,256] f32 -> x1 [B,256,512] bf16 ----------------
__global__ void xpose_kernel(const float* __restrict__ x, bf16_t* __restrict__ x1) {
  __shared__ float tile[32][33];
  int b  = blockIdx.z;
  int s0 = blockIdx.x * 32, c0 = blockIdx.y * 32;
  const float* xb = x + (size_t)b * 512 * 256;
#pragma unroll
  for (int k = 0; k < 4; ++k)
    tile[threadIdx.y + 8 * k][threadIdx.x] =
        xb[(size_t)(s0 + threadIdx.y + 8 * k) * 256 + c0 + threadIdx.x];
  __syncthreads();
  bf16_t* ob = x1 + (size_t)b * 256 * 512;
#pragma unroll
  for (int k = 0; k < 4; ++k)
    ob[(size_t)(c0 + threadIdx.y + 8 * k) * 512 + s0 + threadIdx.x] =
        (bf16_t)tile[threadIdx.x][threadIdx.y + 8 * k];
}

// ---------------- weight [K,N] f32 -> wt [N,K] bf16 (transposed for WMMA B-frag) ----
__global__ void wpose_kernel(const float* __restrict__ w, bf16_t* __restrict__ wt,
                             int K, int N) {
  __shared__ float tile[32][33];
  int k0 = blockIdx.x * 32, n0 = blockIdx.y * 32;
#pragma unroll
  for (int r = 0; r < 4; ++r)
    tile[threadIdx.y + 8 * r][threadIdx.x] =
        w[(size_t)(k0 + threadIdx.y + 8 * r) * N + n0 + threadIdx.x];
  __syncthreads();
#pragma unroll
  for (int r = 0; r < 4; ++r)
    wt[(size_t)(n0 + threadIdx.y + 8 * r) * K + k0 + threadIdx.x] =
        (bf16_t)tile[threadIdx.x][threadIdx.y + 8 * r];
}

// ---------------- generic batched WMMA bf16 GEMM --------------------------------
// C[z] = alpha * A[z] @ Bt[z]^T (+ bias); per-batch M == 256 in every use here.
// A base for batch z: A + (z>>a_nh_shift)*a_super_stride + (z&mask)*a_sub_off
// (K-slice trick so x1 serves all three attention scales).
// Bt is [N,K] row-major (K contiguous) -> B fragment is contiguous LDS reads.
// Staging uses GLOBAL_LOAD_ASYNC_TO_LDS_B128 (ASYNCcnt) so tile k+1 streams into
// LDS via DMA while WMMAs run on tile k; one s_wait_asynccnt+barrier per K-chunk.
template <int FLAGS>
__global__ __launch_bounds__(256)
void gemm_wmma_bf16(const bf16_t* __restrict__ A, const bf16_t* __restrict__ Bt,
                    const float* __restrict__ bias, void* __restrict__ Cout,
                    int K, int lda,
                    long a_super_stride, int a_nh_shift, int a_sub_off,
                    long b_batch_stride, long c_batch_stride, int ldc,
                    float alpha) {
  __shared__ __align__(16) bf16_t sA[2][128][40];  // 32 K-cols + 8 pad
  __shared__ __align__(16) bf16_t sB[2][128][40];

  int z = blockIdx.z;
  const bf16_t* Ab = A + ((long)(z >> a_nh_shift)) * a_super_stride +
                     (long)(z & ((1 << a_nh_shift) - 1)) * a_sub_off;
  const bf16_t* Bb = Bt + (long)z * b_batch_stride;
  int m0 = blockIdx.y * 128;
  int n0 = blockIdx.x * 128;
  int tid = threadIdx.x;

  // staging coords: 256 threads x 2 chunks of 16B cover each 128x32 bf16 tile
  int r0  = tid >> 2;        // 0..63
  int c0q = (tid & 3) * 8;   // 0,8,16,24

  // LDS byte offsets (flat address low 32 bits == LDS offset, ISA 10.2)
  unsigned lA[2][2], lB[2][2];
#pragma unroll
  for (int b = 0; b < 2; ++b)
#pragma unroll
    for (int i = 0; i < 2; ++i) {
      lA[b][i] = (unsigned)(unsigned long long)&sA[b][r0 + 64 * i][c0q];
      lB[b][i] = (unsigned)(unsigned long long)&sB[b][r0 + 64 * i][c0q];
    }
  const bf16_t* gA[2];
  const bf16_t* gB[2];
#pragma unroll
  for (int i = 0; i < 2; ++i) {
    gA[i] = Ab + (long)(m0 + r0 + 64 * i) * lda + c0q;
    gB[i] = Bb + (long)(n0 + r0 + 64 * i) * K + c0q;
  }

  auto stage = [&](int b) {
#pragma unroll
    for (int i = 0; i < 2; ++i) {
      asm volatile("global_load_async_to_lds_b128 %0, %1, off"
                   :: "v"(lA[b][i]), "v"(gA[i]) : "memory");
      asm volatile("global_load_async_to_lds_b128 %0, %1, off"
                   :: "v"(lB[b][i]), "v"(gB[i]) : "memory");
      gA[i] += 32;   // advance one K-chunk
      gB[i] += 32;
    }
  };

  int wid  = tid >> 5;
  int wm   = wid & 3;     // 4 waves along M -> 32 rows each
  int wn   = wid >> 2;    // 2 waves along N -> 64 cols each
  int lane = tid & 31;
  int h    = lane >> 4;   // half-wave
  int ln   = lane & 15;

  v8f acc[2][4] = {};
  int KT = K >> 5;

  stage(0);
  asm volatile("s_wait_asynccnt 0x0" ::: "memory");
  __syncthreads();

  for (int kt = 0; kt < KT; ++kt) {
    int buf = kt & 1;
    if (kt + 1 < KT) stage(buf ^ 1);  // DMA next tile while we compute this one

    // A fragments (16x32 bf16): lane(h,m): elems 0-7 <- K=8h.., elems 8-15 <- K=16+8h..
    Frag32 af[2], bfr[4];
#pragma unroll
    for (int i = 0; i < 2; ++i) {
      int r = wm * 32 + i * 16 + ln;
      af[i].h[0] = *(const v8bf*)&sA[buf][r][8 * h];
      af[i].h[1] = *(const v8bf*)&sA[buf][r][16 + 8 * h];
    }
    // B fragments (32x16): lane(h,n): elems 0-7 <- K=8h.., elems 8-15 <- K=16+8h.., column n (Bt row = n)
#pragma unroll
    for (int j = 0; j < 4; ++j) {
      int r = wn * 64 + j * 16 + ln;
      bfr[j].h[0] = *(const v8bf*)&sB[buf][r][8 * h];
      bfr[j].h[1] = *(const v8bf*)&sB[buf][r][16 + 8 * h];
    }
#pragma unroll
    for (int i = 0; i < 2; ++i)
#pragma unroll
      for (int j = 0; j < 4; ++j)
        acc[i][j] = __builtin_amdgcn_wmma_f32_16x16x32_bf16(
            false, af[i].v, false, bfr[j].v, (short)0, acc[i][j], false, false);

    if (kt + 1 < KT) {
      asm volatile("s_wait_asynccnt 0x0" ::: "memory");  // next tile landed in LDS
      __syncthreads();                                   // visible to all waves
    }
  }

  // Epilogue: C/D layout lane(h,ln): elem e -> row = 8h+e, col = ln
  char* Cb = (char*)Cout;
  constexpr bool obf = (FLAGS & FLAG_OUT_BF16) != 0;
  constexpr bool otr = (FLAGS & FLAG_TRANS) != 0;
  constexpr bool obi = (FLAGS & FLAG_BIAS) != 0;
  long cbase = (long)z * c_batch_stride;
#pragma unroll
  for (int i = 0; i < 2; ++i) {
    int mb = m0 + wm * 32 + i * 16 + 8 * h;
#pragma unroll
    for (int j = 0; j < 4; ++j) {
      int col = n0 + wn * 64 + j * 16 + ln;
      float bcol = obi ? bias[col] : 0.0f;
#pragma unroll
      for (int e = 0; e < 8; ++e) {
        int row = mb + e;
        float val = acc[i][j][e] * alpha + bcol;
        long idx = cbase + (otr ? ((long)col * ldc + row) : ((long)row * ldc + col));
        if constexpr (obf) ((bf16_t*)Cb)[idx] = (bf16_t)val;
        else               ((float*)Cb)[idx]  = val;
      }
    }
  }
}

// ---------------- row softmax over 256 columns, in place -------------------------
__global__ void softmax256_kernel(float* __restrict__ s) {
  float* row = s + (size_t)blockIdx.x * 256;
  int t = threadIdx.x;
  float v = row[t];
  __shared__ float red[256];
  red[t] = v;
  __syncthreads();
  for (int o = 128; o > 0; o >>= 1) {
    if (t < o) red[t] = fmaxf(red[t], red[t + o]);
    __syncthreads();
  }
  float m = red[0];
  __syncthreads();
  float e = __expf(v - m);
  red[t] = e;
  __syncthreads();
  for (int o = 128; o > 0; o >>= 1) {
    if (t < o) red[t] += red[t + o];
    __syncthreads();
  }
  row[t] = e / red[0];
}

// ------- gate: average scale-2/3 sub-batches, softmax over scales, combine -------
__global__ void gate_kernel(const float* __restrict__ s1, const float* __restrict__ s2,
                            const float* __restrict__ s3, bf16_t* __restrict__ attn) {
  size_t i = (size_t)blockIdx.x * blockDim.x + threadIdx.x;  // over 64*256*256
  size_t b = i >> 16;
  size_t r = i & 0xFFFF;
  float a1 = s1[i];
  float a2 = 0.5f * (s2[((b * 2) << 16) + r] + s2[((b * 2 + 1) << 16) + r]);
  float a3 = 0.25f * (s3[((b * 4) << 16) + r] + s3[((b * 4 + 1) << 16) + r] +
                      s3[((b * 4 + 2) << 16) + r] + s3[((b * 4 + 3) << 16) + r]);
  float m  = fmaxf(a1, fmaxf(a2, a3));
  float e1 = __expf(a1 - m), e2 = __expf(a2 - m), e3 = __expf(a3 - m);
  float inv = 1.0f / (e1 + e2 + e3);
  attn[i] = (bf16_t)((a1 * e1 + a2 * e2 + a3 * e3) * inv);
}

extern "C" void kernel_launch(void* const* d_in, const int* in_sizes, int n_in,
                              void* d_out, int out_size, void* d_ws, size_t ws_size,
                              hipStream_t stream) {
  (void)in_sizes; (void)n_in; (void)out_size; (void)ws_size;
  const float* x   = (const float*)d_in[0];
  const float* wq1 = (const float*)d_in[1];  const float* bq1 = (const float*)d_in[2];
  const float* wk1 = (const float*)d_in[3];  const float* bk1 = (const float*)d_in[4];
  const float* wq2 = (const float*)d_in[5];  const float* bq2 = (const float*)d_in[6];
  const float* wk2 = (const float*)d_in[7];  const float* bk2 = (const float*)d_in[8];
  const float* wq3 = (const float*)d_in[9];  const float* bq3 = (const float*)d_in[10];
  const float* wk3 = (const float*)d_in[11]; const float* bk3 = (const float*)d_in[12];
  const float* wv  = (const float*)d_in[13]; const float* bv  = (const float*)d_in[14];
  const float* wo  = (const float*)d_in[15]; const float* bo  = (const float*)d_in[16];

  char* p = (char*)d_ws;
  auto take = [&](size_t bytes) -> char* {
    char* r = p;
    p += (bytes + 255) & ~(size_t)255;
    return r;
  };

  bf16_t* x1   = (bf16_t*)take((size_t)64 * 256 * 512 * 2);
  bf16_t* wq1T = (bf16_t*)take((size_t)128 * 512 * 2);
  bf16_t* wk1T = (bf16_t*)take((size_t)128 * 512 * 2);
  bf16_t* wq2T = (bf16_t*)take((size_t)128 * 256 * 2);
  bf16_t* wk2T = (bf16_t*)take((size_t)128 * 256 * 2);
  bf16_t* wq3T = (bf16_t*)take((size_t)128 * 128 * 2);
  bf16_t* wk3T = (bf16_t*)take((size_t)128 * 128 * 2);
  bf16_t* wvT  = (bf16_t*)take((size_t)512 * 512 * 2);
  bf16_t* woT  = (bf16_t*)take((size_t)512 * 512 * 2);
  bf16_t* q1   = (bf16_t*)take((size_t)64  * 256 * 128 * 2);
  bf16_t* k1   = (bf16_t*)take((size_t)64  * 256 * 128 * 2);
  bf16_t* q2   = (bf16_t*)take((size_t)128 * 256 * 128 * 2);
  bf16_t* k2   = (bf16_t*)take((size_t)128 * 256 * 128 * 2);
  bf16_t* q3   = (bf16_t*)take((size_t)256 * 256 * 128 * 2);
  bf16_t* k3   = (bf16_t*)take((size_t)256 * 256 * 128 * 2);
  float*  s1   = (float*) take((size_t)64  * 256 * 256 * 4);
  float*  s2   = (float*) take((size_t)128 * 256 * 256 * 4);
  float*  s3   = (float*) take((size_t)256 * 256 * 256 * 4);
  // reuse dead buffers after the scores stage
  bf16_t* attn = q1;  // needs 8.4 MB, q1+k1 region = 8.4 MB
  bf16_t* vT   = q3;  // 16.8 MB
  bf16_t* tbuf = k3;  // 16.8 MB

  dim3 tb32(32, 8);
  xpose_kernel<<<dim3(16, 8, 64), tb32, 0, stream>>>(x, x1);
  wpose_kernel<<<dim3(16, 4), tb32, 0, stream>>>(wq1, wq1T, 512, 128);
  wpose_kernel<<<dim3(16, 4), tb32, 0, stream>>>(wk1, wk1T, 512, 128);
  wpose_kernel<<<dim3(8, 4),  tb32, 0, stream>>>(wq2, wq2T, 256, 128);
  wpose_kernel<<<dim3(8, 4),  tb32, 0, stream>>>(wk2, wk2T, 256, 128);
  wpose_kernel<<<dim3(4, 4),  tb32, 0, stream>>>(wq3, wq3T, 128, 128);
  wpose_kernel<<<dim3(4, 4),  tb32, 0, stream>>>(wk3, wk3T, 128, 128);
  wpose_kernel<<<dim3(16, 16), tb32, 0, stream>>>(wv, wvT, 512, 512);
  wpose_kernel<<<dim3(16, 16), tb32, 0, stream>>>(wo, woT, 512, 512);

  const float rsH = 0.08838834764831845f;  // 1/sqrt(128)

  // Q/K projections, all three scales (A = K-slices of x1 rows)
  gemm_wmma_bf16<FLAG_OUT_BF16 | FLAG_BIAS><<<dim3(1, 2, 64), dim3(256), 0, stream>>>(
      x1, wq1T, bq1, q1, 512, 512, 131072L, 0, 0, 0L, 32768L, 128, 1.0f);
  gemm_wmma_bf16<FLAG_OUT_BF16 | FLAG_BIAS><<<dim3(1, 2, 64), dim3(256), 0, stream>>>(
      x1, wk1T, bk1, k1, 512, 512, 131072L, 0, 0, 0L, 32768L, 128, 1.0f);
  gemm_wmma_bf16<FLAG_OUT_BF16 | FLAG_BIAS><<<dim3(1, 2, 128), dim3(256), 0, stream>>>(
      x1, wq2T, bq2, q2, 256, 512, 131072L, 1, 256, 0L, 32768L, 128, 1.0f);
  gemm_wmma_bf16<FLAG_OUT_BF16 | FLAG_BIAS><<<dim3(1, 2, 128), dim3(256), 0, stream>>>(
      x1, wk2T, bk2, k2, 256, 512, 131072L, 1, 256, 0L, 32768L, 128, 1.0f);
  gemm_wmma_bf16<FLAG_OUT_BF16 | FLAG_BIAS><<<dim3(1, 2, 256), dim3(256), 0, stream>>>(
      x1, wq3T, bq3, q3, 128, 512, 131072L, 2, 128, 0L, 32768L, 128, 1.0f);
  gemm_wmma_bf16<FLAG_OUT_BF16 | FLAG_BIAS><<<dim3(1, 2, 256), dim3(256), 0, stream>>>(
      x1, wk3T, bk3, k3, 128, 512, 131072L, 2, 128, 0L, 32768L, 128, 1.0f);

  // scores = q @ k^T / sqrt(H); k already is the [N,K] B^T layout
  gemm_wmma_bf16<0><<<dim3(2, 2, 64), dim3(256), 0, stream>>>(
      q1, k1, (const float*)nullptr, s1, 128, 128, 32768L, 0, 0, 32768L, 65536L, 256, rsH);
  gemm_wmma_bf16<0><<<dim3(2, 2, 128), dim3(256), 0, stream>>>(
      q2, k2, (const float*)nullptr, s2, 128, 128, 32768L, 0, 0, 32768L, 65536L, 256, rsH);
  gemm_wmma_bf16<0><<<dim3(2, 2, 256), dim3(256), 0, stream>>>(
      q3, k3, (const float*)nullptr, s3, 128, 128, 32768L, 0, 0, 32768L, 65536L, 256, rsH);

  softmax256_kernel<<<dim3(64 * 256), dim3(256), 0, stream>>>(s1);
  softmax256_kernel<<<dim3(128 * 256), dim3(256), 0, stream>>>(s2);
  softmax256_kernel<<<dim3(256 * 256), dim3(256), 0, stream>>>(s3);

  gate_kernel<<<dim3(16384), dim3(256), 0, stream>>>(s1, s2, s3, attn);

  // v^T = (x1 @ wv + bv)^T, stored [B,512,256] so it is the B^T layout for attn@v
  gemm_wmma_bf16<FLAG_OUT_BF16 | FLAG_TRANS | FLAG_BIAS>
      <<<dim3(4, 2, 64), dim3(256), 0, stream>>>(
      x1, wvT, bv, vT, 512, 512, 131072L, 0, 0, 0L, 131072L, 256, 1.0f);

  // t = attn @ v   (Bt = v^T, per-batch)
  gemm_wmma_bf16<FLAG_OUT_BF16><<<dim3(4, 2, 64), dim3(256), 0, stream>>>(
      attn, vT, (const float*)nullptr, tbuf, 256, 256, 65536L, 0, 0, 131072L, 131072L, 512, 1.0f);

  // out = (t @ wo + bo), stored transposed fp32 -> d_out [B, P, C]
  gemm_wmma_bf16<FLAG_TRANS | FLAG_BIAS><<<dim3(4, 2, 64), dim3(256), 0, stream>>>(
      tbuf, woT, bo, d_out, 512, 512, 131072L, 0, 0, 0L, 131072L, 256, 1.0f);
}
